// LossSSIM_45251775430674
// MI455X (gfx1250) — matrix operand order repaired and checked by reading, not compiled, not run on hardware
//
#include <hip/hip_runtime.h>

// ---------------------------------------------------------------------------
// SSIM loss, fused, for MI455X (gfx1250, wave32, WMMA).
//
//  - One wave (32 threads) per 16x16 conv-output tile. Grid = (32,32,96).
//  - Separable 11-tap Gaussian conv done as two banded GEMMs on the matrix
//    pipe: v_wmma_f32_16x16x32_f16 with static banded weight operands.
//  - f16 conv inputs, f32 accumulation. Bilinear resize done exactly with
//    integer arithmetic (pos = Y*383/511 via int div/mod).
//  - Deterministic: per-block partials + fixed-order reduction (no atomics).
// ---------------------------------------------------------------------------

typedef __attribute__((ext_vector_type(16))) _Float16 v16h;
typedef __attribute__((ext_vector_type(8)))  float    v8f;
typedef unsigned int uint;

#define NTX 32
#define NTY 32
#define NIMG 96
#define NTILES (NTX * NTY * NIMG)

__device__ __forceinline__ float gaussw(int d, float invS) {
    // normalized 1D Gaussian tap, d in [0,10], center 5, sigma 1.5
    float x = (float)(d - 5);
    return __expf(-x * x * (1.0f / 4.5f)) * invS;
}

union H2U { uint u; _Float16 h[2]; };

__global__ __launch_bounds__(32)
void ssim_tile_kernel(const float* __restrict__ inp,   // [96][384][384]
                      const float* __restrict__ tgt,   // [96][512][512]
                      float* __restrict__ partial) {   // [NTILES]
    __shared__ __align__(16) _Float16 s_maps[5][32][36]; // 5 product maps, f16, padded
    __shared__ __align__(16) float    s_H[32][17];       // h-pass result bounce buffer

    const int lane = threadIdx.x;          // 0..31, one full wave
    const int tx = blockIdx.x, ty = blockIdx.y, img = blockIdx.z;
    const int oy0 = ty * 16, ox0 = tx * 16;          // conv-output tile origin
    const float* ip = inp + (size_t)img * (384 * 384);
    const float* tp = tgt + (size_t)img * (512 * 512);

    // Prefetch target halo rows for this tile (global_prefetch_b8).
    if (lane < 26) {
        int Y = oy0 + lane;
        if (Y < 512 && ox0 < 512) __builtin_prefetch(tp + Y * 512 + ox0, 0, 0);
    }

    // ---- zero the LDS maps (halo/pad region must be 0, never NaN) ----------
    for (int i = lane; i < 5 * 32 * 36 / 2; i += 32) ((uint*)s_maps)[i] = 0u;
    __syncthreads();

    // ---- phase A: bilinear upsample + target + 5 product maps (26x26) ------
    for (int e = lane; e < 26 * 26; e += 32) {
        int r = e / 26, c = e - r * 26;
        int Y = oy0 + r, X = ox0 + c;
        if (Y < 512 && X < 512) {
            float t = tp[Y * 512 + X];
            // align_corners: pos = Y * 383 / 511 (exact integer split)
            int ry = Y * 383; int iy0 = ry / 511; float wy = (float)(ry - iy0 * 511) * (1.0f / 511.0f);
            int rx = X * 383; int ix0 = rx / 511; float wx = (float)(rx - ix0 * 511) * (1.0f / 511.0f);
            int iy1 = iy0 + 1; if (iy1 > 383) iy1 = 383;
            int ix1 = ix0 + 1; if (ix1 > 383) ix1 = 383;
            float a00 = ip[iy0 * 384 + ix0], a01 = ip[iy0 * 384 + ix1];
            float a10 = ip[iy1 * 384 + ix0], a11 = ip[iy1 * 384 + ix1];
            float top = a00 + wx * (a01 - a00);
            float bot = a10 + wx * (a11 - a10);
            float v1  = top + wy * (bot - top);
            s_maps[0][r][c] = (_Float16)v1;
            s_maps[1][r][c] = (_Float16)t;
            s_maps[2][r][c] = (_Float16)(v1 * v1);
            s_maps[3][r][c] = (_Float16)(t * t);
            s_maps[4][r][c] = (_Float16)(v1 * t);
        }
    }
    __syncthreads();

    // ---- Gaussian weights ---------------------------------------------------
    float S = 0.0f;
#pragma unroll
    for (int j = 0; j < 11; ++j) { float x = (float)(j - 5); S += __expf(-x * x * (1.0f / 4.5f)); }
    const float invS = 1.0f / S;

    const int n    = lane & 15;              // B/C/D column this lane owns
    const int mb   = (lane < 16) ? 0 : 8;    // C/D row base for this lane
    const int kb16 = (lane < 16) ? 0 : 16;   // unused; B's k base for this lane is mb

    // B operand: W_b[k, x] = g[k - x]  (32x16 banded, static)
    v16h bw;
#pragma unroll
    for (int h = 0; h < 16; ++h) {
        int d = (h & 7) + ((h & 8) << 1) + mb - n;
        bw[h] = (_Float16)((d >= 0 && d <= 10) ? gaussw(d, invS) : 0.0f);
    }
    // A operand: W_a[y, k] = g[k - y]  (16x32 banded, static)
    const int M = lane & 15;
    v16h aw;
#pragma unroll
    for (int h = 0; h < 16; ++h) {
        int K = (h & 7) + ((h < 8) ? 0 : 16) + ((lane < 16) ? 0 : 8);
        int d = K - M;
        aw[h] = (_Float16)((d >= 0 && d <= 10) ? gaussw(d, invS) : 0.0f);
    }

    // ---- phase B: per map, 2 h-pass WMMAs + 1 v-pass WMMA -------------------
    v8f outs[5];
#pragma unroll
    for (int m = 0; m < 5; ++m) {
        // A operands: image rows in the 16-bit A layout (dword pair reads).
        v16h a0, a1;
#pragma unroll
        for (int h = 0; h < 16; h += 2) {
            int K = (h & 7) + ((h < 8) ? 0 : 16) + ((lane < 16) ? 0 : 8); // even
            H2U u0, u1;
            u0.u = *(const uint*)&s_maps[m][M][K];
            u1.u = *(const uint*)&s_maps[m][M + 16][K];
            a0[h] = u0.h[0]; a0[h + 1] = u0.h[1];
            a1[h] = u1.h[0]; a1[h + 1] = u1.h[1];
        }
        v8f cz = {};
        v8f h0 = __builtin_amdgcn_wmma_f32_16x16x32_f16(false, a0, false, bw, (short)0, cz, false, false);
        v8f h1 = __builtin_amdgcn_wmma_f32_16x16x32_f16(false, a1, false, bw, (short)0, cz, false, false);

        __syncthreads();                        // previous map's s_H reads done
#pragma unroll
        for (int j = 0; j < 8; ++j) {
            s_H[mb + j][n]      = h0[j];        // rows 0..15
            s_H[16 + mb + j][n] = h1[j];        // rows 16..31 (26..31 are zero)
        }
        __syncthreads();

        v16h bh;                                // B operand: H as f16, 32x16
#pragma unroll
        for (int h = 0; h < 16; ++h) bh[h] = (_Float16)s_H[(h & 7) + ((h & 8) << 1) + mb][n];

        outs[m] = __builtin_amdgcn_wmma_f32_16x16x32_f16(false, aw, false, bh, (short)0, cz, false, false);
    }

    // ---- SSIM map + masked accumulation (C/D layout shared by all 5) --------
    const float C1f = 0.0001f;   // (0.01)^2
    const float C2f = 0.0009f;   // (0.03)^2
    float acc = 0.0f;
#pragma unroll
    for (int j = 0; j < 8; ++j) {
        float mu1 = outs[0][j], mu2 = outs[1][j];
        float e11 = outs[2][j], e22 = outs[3][j], e12 = outs[4][j];
        float mu1s = mu1 * mu1, mu2s = mu2 * mu2, mu12 = mu1 * mu2;
        float s1 = e11 - mu1s, s2 = e22 - mu2s, s12 = e12 - mu12;
        float v1 = 2.0f * s12 + C2f;
        float v2 = s1 + s2 + C2f;
        float num = (2.0f * mu12 + C1f) * v1;
        float den = (mu1s + mu2s + C1f) * v2;
        float ss = num / den;
        int oy = oy0 + mb + j, ox = ox0 + n;
        if (oy < 502 && ox < 502) acc += ss;
    }

    // deterministic wave reduction through LDS, one partial per block
    __syncthreads();
    ((float*)s_H)[lane] = acc;
    __syncthreads();
    if (lane == 0) {
        float s = 0.0f;
#pragma unroll
        for (int i = 0; i < 32; ++i) s += ((float*)s_H)[i];
        partial[(blockIdx.z * NTY + blockIdx.y) * NTX + blockIdx.x] = s;
    }
}

__global__ __launch_bounds__(256)
void ssim_finalize_kernel(const float* __restrict__ partial, float* __restrict__ out) {
    __shared__ double red[256];
    double s = 0.0;
    for (int i = threadIdx.x; i < NTILES; i += 256) s += (double)partial[i];
    red[threadIdx.x] = s;
    __syncthreads();
    for (int k = 128; k > 0; k >>= 1) {
        if ((int)threadIdx.x < k) red[threadIdx.x] += red[threadIdx.x + k];
        __syncthreads();
    }
    if (threadIdx.x == 0) {
        double mean = red[0] / (double)(96.0 * 502.0 * 502.0);
        double l = 1.0 - mean;
        if (l < 0.0) l = 0.0;
        if (l > 1.0) l = 1.0;
        out[0] = (float)l;
    }
}

extern "C" void kernel_launch(void* const* d_in, const int* in_sizes, int n_in,
                              void* d_out, int out_size, void* d_ws, size_t ws_size,
                              hipStream_t stream) {
    (void)in_sizes; (void)n_in; (void)out_size; (void)ws_size;
    const float* inp = (const float*)d_in[0];   // [32,3,384,384] f32
    const float* tgt = (const float*)d_in[1];   // [32,3,512,512] f32
    float* partial = (float*)d_ws;              // NTILES * 4 = 384 KB scratch

    dim3 grid(NTX, NTY, NIMG);
    ssim_tile_kernel<<<grid, 32, 0, stream>>>(inp, tgt, partial);
    ssim_finalize_kernel<<<1, 256, 0, stream>>>(partial, (float*)d_out);
}
